// QuantizedLinear_67843303408008
// MI455X (gfx1250) — compile-verified, matrix operand order checked by reading, not run on hardware
//
#include <hip/hip_runtime.h>

typedef __attribute__((ext_vector_type(16))) _Float16 v16h;
typedef __attribute__((ext_vector_type(2)))  _Float16 h2;
typedef __attribute__((ext_vector_type(8)))  float    v8f;
typedef __attribute__((ext_vector_type(4)))  int      v4i;
typedef __attribute__((ext_vector_type(4)))  float    v4f;

#define IN_F   4096
#define OUT_F  11008
#define GPR    32        // groups per output row (4096/128)
#define NGRP   344       // N-groups: 1376 tiles / 4 tiles (32 outputs) per block

// Decode 8 packed int4-pair bytes (each in its own i32) to a 16-half B fragment.
// Magic trick: 0x6400|n is exactly (1024+n) in f16; (t - 1032) == nibble-8 exactly.
__device__ __forceinline__ v16h decode16(v4i wa, v4i wb, h2 sh) {
    union { v16h v; h2 p[8]; } r;
    const h2 bias = {(_Float16)(-1032.0f), (_Float16)(-1032.0f)};
    #pragma unroll
    for (int j = 0; j < 4; ++j) {
        unsigned ua = (unsigned)wa[j];
        unsigned ub = (unsigned)wb[j];
        unsigned ta = ((ua | (ua << 12)) & 0x000F000Fu) | 0x64006400u;
        unsigned tb = ((ub | (ub << 12)) & 0x000F000Fu) | 0x64006400u;
        r.p[j]     = (__builtin_bit_cast(h2, ta) + bias) * sh;   // K = 2j, 2j+1
        r.p[4 + j] = (__builtin_bit_cast(h2, tb) + bias) * sh;   // K = 16+2j, 16+2j+1
    }
    return r.v;
}

// Block: 256 thr = 8 waves = one N-group (32 output features = 4 tiles) x 8 K-splits.
// Wave: two 16x16x32 WMMAs per iteration sharing one A fragment; K range = 512.
__global__ __launch_bounds__(256) void q4_gemm_wmma(
    const float* __restrict__ x,
    const int*   __restrict__ wq,
    const float* __restrict__ sc,
    float* __restrict__ out)
{
    __shared__ v8f red[7][32][2];   // 14 KB partial-sum buffer (waves 1..7)

    const int tid  = threadIdx.x;
    const int wv   = tid >> 5;         // wave 0..7 == K-split index
    const int lane = tid & 31;
    const int ng   = blockIdx.x;       // N-group 0..343 (32 output features)

    const int hiH  = (lane >> 4) & 1;  // 0: lanes 0-15, 1: lanes 16-31
    const int n    = lane & 15;        // column within 16-wide WMMA (== A row M)
    const int t0   = ng * 4 + (n >> 3);// tile for first WMMA; second is t0+2
    const int m    = n & 7;            // row within tile
    const int M    = n;                // A-matrix row held by this lane

    const float* xrow = x + (size_t)M * IN_F;
    v8f c0 = {}, c1 = {};

    for (int gj = 0; gj < 4; ++gj) {
        const int g = wv * 4 + gj;                             // global group
        const size_t rowbase = ((size_t)t0 * GPR + g) * 8 + m;
        const float s0 = sc[rowbase];
        const float s1 = sc[rowbase + 2 * GPR * 8];            // tile t0+2
        const h2 sh0 = {(_Float16)s0, (_Float16)s0};
        const h2 sh1 = {(_Float16)s1, (_Float16)s1};
        const int* wrow = wq + rowbase * 64 + hiH * 4;
        __builtin_prefetch(wrow + 512, 0, 1);                  // next group, tile 0
        __builtin_prefetch(wrow + 32768 + 512, 0, 1);          // next group, tile 1
        const float* xp = xrow + g * 128 + hiH * 8;

        #pragma unroll
        for (int c32 = 0; c32 < 4; ++c32) {
            const int h = c32 * 16;
            // packed weights: 4 coalesced b128 loads, one base + imm offsets
            v4i wa0 = *(const v4i*)(wrow + h);
            v4i wb0 = *(const v4i*)(wrow + h + 8);
            v4i wa1 = *(const v4i*)(wrow + h + 32768);
            v4i wb1 = *(const v4i*)(wrow + h + 32768 + 8);

            // A fragment: x rows f32->f16; rows >= 8 are zero pad
            v16h a = {};
            if (M < 8) {
                v4f x0 = *(const v4f*)(xp + c32 * 32);
                v4f x1 = *(const v4f*)(xp + c32 * 32 + 4);
                v4f x2 = *(const v4f*)(xp + c32 * 32 + 16);
                v4f x3 = *(const v4f*)(xp + c32 * 32 + 20);
                #pragma unroll
                for (int j = 0; j < 4; ++j) {
                    a[j]      = (_Float16)x0[j];
                    a[4 + j]  = (_Float16)x1[j];
                    a[8 + j]  = (_Float16)x2[j];
                    a[12 + j] = (_Float16)x3[j];
                }
            }

            v16h b0 = decode16(wa0, wb0, sh0);
            v16h b1 = decode16(wa1, wb1, sh1);

            // EXEC is all-ones here (divergence above has reconverged)
            c0 = __builtin_amdgcn_wmma_f32_16x16x32_f16(
                     false, a, false, b0, (short)0, c0, false, false);
            c1 = __builtin_amdgcn_wmma_f32_16x16x32_f16(
                     false, a, false, b1, (short)0, c1, false, false);
        }
    }

    // 8-way split-K reduction through LDS
    if (wv != 0) {
        red[wv - 1][lane][0] = c0;
        red[wv - 1][lane][1] = c1;
    }
    __syncthreads();
    if (wv == 0) {
        #pragma unroll
        for (int ww = 0; ww < 7; ++ww) {
            c0 += red[ww][lane][0];
            c1 += red[ww][lane][1];
        }
        // C layout: VGPR r, lanes 0-15 -> row M=r, col N=lane (rows 0..7 = batch)
        if (lane < 16) {
            float* orow = out + (size_t)ng * 32 + n;
            #pragma unroll
            for (int r = 0; r < 8; ++r) {
                orow[(size_t)r * OUT_F]      = c0[r];   // features ng*32 + n
                orow[(size_t)r * OUT_F + 16] = c1[r];   // features ng*32 + 16 + n
            }
        }
    }
}

extern "C" void kernel_launch(void* const* d_in, const int* in_sizes, int n_in,
                              void* d_out, int out_size, void* d_ws, size_t ws_size,
                              hipStream_t stream) {
    const float* x  = (const float*)d_in[0];   // (8, 4096) f32
    const int*   wq = (const int*)d_in[1];     // (1376, 32, 8, 64) packed bytes in i32
    const float* sc = (const float*)d_in[2];   // (1376, 32, 8) f32
    float* out = (float*)d_out;                // (8, 11008) f32

    q4_gemm_wmma<<<dim3(NGRP), 256, 0, stream>>>(x, wq, sc, out);
}
